// ASP_OC_GAP_SECAMNetHead_50646254355105
// MI455X (gfx1250) — compile-verified, matrix operand order checked by reading, not run on hardware
//
#include <hip/hip_runtime.h>
#include <hip/hip_bf16.h>
#include <stdint.h>
#include <stddef.h>

// ---------------------------------------------------------------------------
// CDNA5 (gfx1250) implementation of the ASP_OC_GAP_SECAMNet head.
// All matmul-shaped work (1x1 convs, dilated 3x3 convs as implicit GEMM,
// PAM/GCAM attention batched GEMMs, fuse convs) runs on
// v_wmma_f32_16x16x32_bf16 (bf16 operands, f32 accumulate).
// The GEMM/conv hot loops are software-pipelined with double-buffered LDS:
// global loads of tile k+1 are issued before the WMMA block of tile k.
// All GEMM/conv tile dims are multiples of 64 (M,N) / 32 (K) by construction,
// so the hot loops carry no bounds checks at all.
// ---------------------------------------------------------------------------

typedef __bf16 bf16;
typedef __attribute__((ext_vector_type(8)))  __bf16 v8bf;
typedef __attribute__((ext_vector_type(16))) __bf16 v16bf;
typedef __attribute__((ext_vector_type(8)))  float  v8f;

union BFrag { v16bf v; v8bf h[2]; };

struct Epi {
    const float* scale;      // per-row (channel) scale  (BN)
    const float* bias;       // per-row bias
    const float* gamma;      // device scalar multiplier (attention gamma)
    const float* res;        // residual added AFTER activation
    long long resBatch; int ldRes;
    int relu;
    float* outF; long long outFBatch; int ldF;
    bf16*  outB; long long outBBatch; int ldBq;
};

__device__ __forceinline__ void epi_store(const Epi& e, int bz, int m, int n, float v) {
    if (e.scale)      v = v * e.scale[m] + (e.bias ? e.bias[m] : 0.f);
    else if (e.bias)  v += e.bias[m];
    if (e.gamma)      v *= e.gamma[0];
    if (e.relu)       v = v > 0.f ? v : 0.f;
    if (e.res)        v += e.res[(long long)bz * e.resBatch + (long long)m * e.ldRes + n];
    if (e.outF) e.outF[(long long)bz * e.outFBatch + (long long)m * e.ldF  + n] = v;
    if (e.outB) e.outB[(long long)bz * e.outBBatch + (long long)m * e.ldBq + n] = (bf16)v;
}

#define LDSW 40   // padded bf16 row stride (80B: keeps 16B-aligned v8bf chunks)

// ------------------------- generic batched WMMA GEMM -----------------------
// C[b] = A[b] x B[b]; A is MxK (transA: A(m,k)=A[k*ldA+m]),
// B is KxN (transB: B(k,n)=B[n*ldB+k]).
// 64x64 block tile, 128 threads (4 waves), each wave owns a 32x32 C tile
// (4 wmma per K-step of 32).  Double-buffered LDS software pipeline.
// Requires M%64==0, N%64==0, K%32==0.
__global__ __launch_bounds__(128)
void gemm_wmma(const bf16* __restrict__ A, long long aBatch, int ldA, int transA,
               const bf16* __restrict__ Bm, long long bBatch, int ldB, int transB,
               int M, int N, int K, Epi e) {
    __shared__ __attribute__((aligned(16))) bf16 AsB[2][64 * LDSW];
    __shared__ __attribute__((aligned(16))) bf16 BsB[2][64 * LDSW];
    const int tid = threadIdx.x, lane = tid & 31, wave = tid >> 5;
    const int bz = blockIdx.z;
    const int m0 = blockIdx.y * 64, n0 = blockIdx.x * 64;
    const bf16* Ab = A  + (long long)bz * aBatch;
    const bf16* Bb = Bm + (long long)bz * bBatch;

    // per-thread staging coordinates + streaming pointers
    int ar[2], ac[2];                    // A: (row,kchunk) or trans (k,mchunk)
    const bf16* pA[2]; long long stepA;
    if (!transA) {
        for (int s = 0; s < 2; ++s) {
            int vt = tid + s * 128;
            ar[s] = vt >> 2; ac[s] = (vt & 3) * 8;
            pA[s] = Ab + (long long)(m0 + ar[s]) * ldA + ac[s];
        }
        stepA = 32;
    } else {
        for (int s = 0; s < 2; ++s) {
            int vt = tid + s * 128;
            ar[s] = vt >> 3; ac[s] = (vt & 7) * 8;
            pA[s] = Ab + (long long)ar[s] * ldA + (m0 + ac[s]);
        }
        stepA = 32LL * ldA;
    }
    int br[2], bc[2];                    // B: (k,nchunk) or trans (n,kchunk)
    const bf16* pB[2]; long long stepB;
    if (!transB) {
        for (int s = 0; s < 2; ++s) {
            int vt = tid + s * 128;
            br[s] = vt >> 3; bc[s] = (vt & 7) * 8;
            pB[s] = Bb + (long long)br[s] * ldB + (n0 + bc[s]);
        }
        stepB = 32LL * ldB;
    } else {
        for (int s = 0; s < 2; ++s) {
            int vt = tid + s * 128;
            br[s] = vt >> 2; bc[s] = (vt & 3) * 8;
            pB[s] = Bb + (long long)(n0 + br[s]) * ldB + bc[s];
        }
        stepB = 32;
    }

    v8f acc00 = {}; v8f acc01 = {}; v8f acc10 = {}; v8f acc11 = {};
    const int mh = (wave & 1) * 32, nh = (wave >> 1) * 32;
    const int fr = lane & 15, kb = (lane < 16) ? 0 : 8;

    v8bf ra[2], rb[2];
    // ---- prologue: stage tile 0 into buffer 0 ----
    for (int s = 0; s < 2; ++s) { ra[s] = *(const v8bf*)pA[s]; pA[s] += stepA; }
    for (int s = 0; s < 2; ++s) { rb[s] = *(const v8bf*)pB[s]; pB[s] += stepB; }
    if (!transA) {
        for (int s = 0; s < 2; ++s) *(v8bf*)&AsB[0][ar[s] * LDSW + ac[s]] = ra[s];
    } else {
        for (int s = 0; s < 2; ++s)
            for (int i = 0; i < 8; ++i) AsB[0][(ac[s] + i) * LDSW + ar[s]] = ra[s][i];
    }
    if (!transB) {
        for (int s = 0; s < 2; ++s)
            for (int i = 0; i < 8; ++i) BsB[0][(bc[s] + i) * LDSW + br[s]] = rb[s][i];
    } else {
        for (int s = 0; s < 2; ++s) *(v8bf*)&BsB[0][br[s] * LDSW + bc[s]] = rb[s];
    }

    const int steps = K >> 5;
    int p = 0;
    for (int it = 0; it < steps; ++it) {
        const bool hasNext = (it + 1 < steps);
        if (hasNext) {   // issue next-tile global loads; wait lands after compute
            for (int s = 0; s < 2; ++s) { ra[s] = *(const v8bf*)pA[s]; pA[s] += stepA; }
            for (int s = 0; s < 2; ++s) { rb[s] = *(const v8bf*)pB[s]; pB[s] += stepB; }
            __builtin_prefetch(pA[0], 0, 2);   // global_prefetch_b8, next+1 tile
        }
        __syncthreads();   // buf[p] fully staged by all waves
        const bf16* As = AsB[p];
        const bf16* Bs = BsB[p];
        BFrag a0, a1, b0, b1;
        a0.h[0] = *(const v8bf*)&As[(mh + fr) * LDSW + kb];
        a0.h[1] = *(const v8bf*)&As[(mh + fr) * LDSW + kb + 16];
        a1.h[0] = *(const v8bf*)&As[(mh + 16 + fr) * LDSW + kb];
        a1.h[1] = *(const v8bf*)&As[(mh + 16 + fr) * LDSW + kb + 16];
        b0.h[0] = *(const v8bf*)&Bs[(nh + fr) * LDSW + kb];
        b0.h[1] = *(const v8bf*)&Bs[(nh + fr) * LDSW + kb + 16];
        b1.h[0] = *(const v8bf*)&Bs[(nh + 16 + fr) * LDSW + kb];
        b1.h[1] = *(const v8bf*)&Bs[(nh + 16 + fr) * LDSW + kb + 16];
        acc00 = __builtin_amdgcn_wmma_f32_16x16x32_bf16(false, a0.v, false, b0.v,
                                                        (short)0, acc00, false, false);
        acc01 = __builtin_amdgcn_wmma_f32_16x16x32_bf16(false, a0.v, false, b1.v,
                                                        (short)0, acc01, false, false);
        acc10 = __builtin_amdgcn_wmma_f32_16x16x32_bf16(false, a1.v, false, b0.v,
                                                        (short)0, acc10, false, false);
        acc11 = __builtin_amdgcn_wmma_f32_16x16x32_bf16(false, a1.v, false, b1.v,
                                                        (short)0, acc11, false, false);
        if (hasNext) {   // store next tile into the other buffer (no readers now)
            bf16* Asn = AsB[p ^ 1];
            bf16* Bsn = BsB[p ^ 1];
            if (!transA) {
                for (int s = 0; s < 2; ++s) *(v8bf*)&Asn[ar[s] * LDSW + ac[s]] = ra[s];
            } else {
                for (int s = 0; s < 2; ++s)
                    for (int i = 0; i < 8; ++i) Asn[(ac[s] + i) * LDSW + ar[s]] = ra[s][i];
            }
            if (!transB) {
                for (int s = 0; s < 2; ++s)
                    for (int i = 0; i < 8; ++i) Bsn[(bc[s] + i) * LDSW + br[s]] = rb[s][i];
            } else {
                for (int s = 0; s < 2; ++s) *(v8bf*)&Bsn[br[s] * LDSW + bc[s]] = rb[s];
            }
        }
        p ^= 1;
    }
    for (int q = 0; q < 4; ++q) {
        v8f av = (q == 0) ? acc00 : (q == 1) ? acc01 : (q == 2) ? acc10 : acc11;
        int mb = mh + ((q >> 1) & 1) * 16;
        int n = n0 + nh + (q & 1) * 16 + fr;
        for (int r = 0; r < 8; ++r) {
            int m = m0 + mb + ((lane < 16) ? r : r + 8);
            epi_store(e, bz, m, n, av[r]);
        }
    }
}

// -------------------- 3x3 conv (dilated / strided) as implicit GEMM --------
// Requires Cout%64==0, Cin%32==0, Hout*Wout%64==0, Wout a power of two.
// Double-buffered pipeline over the flattened (tap, k0) iteration space.
__global__ __launch_bounds__(128)
void conv3x3_wmma(const bf16* __restrict__ X, const bf16* __restrict__ Wp,
                  int Cin, int Cout, int Hin, int Win, int Wout, int wShift,
                  int stride, int pad, int dil, Epi e) {
    __shared__ __attribute__((aligned(16))) bf16 AsB[2][64 * LDSW];
    __shared__ __attribute__((aligned(16))) bf16 BsB[2][64 * LDSW];
    const int tid = threadIdx.x, lane = tid & 31, wave = tid >> 5;
    const int bz = blockIdx.z;
    const int m0 = blockIdx.y * 64, n0 = blockIdx.x * 64;
    const int HWin = Hin * Win;
    const bf16* Xb = X + (long long)bz * Cin * HWin;

    // staging coordinates
    int ar[2], ac[2], bk[2], bn[2];
    for (int s = 0; s < 2; ++s) {
        int vt = tid + s * 128;
        ar[s] = vt >> 2; ac[s] = (vt & 3) * 8;   // A: row, kchunk
        bk[s] = vt >> 3; bn[s] = (vt & 7) * 8;   // B: k, nchunk
    }

    v8f acc00 = {}; v8f acc01 = {}; v8f acc10 = {}; v8f acc11 = {};
    const int mh = (wave & 1) * 32, nh = (wave >> 1) * 32;
    const int fr = lane & 15, kb = (lane < 16) ? 0 : 8;

    const int steps = 9 * (Cin >> 5);
    int tap = 0, k0 = 0;                 // coordinates of the tile being LOADED
    v8bf ra[2];
    bf16 rbv[16];

    // tile loader (tap,k0) -> registers
    auto load_tile = [&]() {
        const bf16* At = Wp + (long long)tap * Cout * Cin;
        for (int s = 0; s < 2; ++s)
            ra[s] = *(const v8bf*)&At[(long long)(m0 + ar[s]) * Cin + (k0 + ac[s])];
        const int ky = tap / 3, kx = tap - ky * 3;
        const int oyOff = ky * dil - pad, oxOff = kx * dil - pad;
        for (int s = 0; s < 2; ++s) {
            const bf16* Xc = Xb + (long long)(k0 + bk[s]) * HWin;
            for (int i = 0; i < 8; ++i) {
                int n = n0 + bn[s] + i;
                int oy = n >> wShift, ox = n & ((1 << wShift) - 1);
                int iy = oy * stride + oyOff;
                int ix = ox * stride + oxOff;
                bool ok = (iy >= 0) & (iy < Hin) & (ix >= 0) & (ix < Win);
                int iyc = iy < 0 ? 0 : (iy >= Hin ? Hin - 1 : iy);
                int ixc = ix < 0 ? 0 : (ix >= Win ? Win - 1 : ix);
                bf16 val = Xc[iyc * Win + ixc];
                rbv[s * 8 + i] = ok ? val : (bf16)0.f;
            }
        }
        k0 += 32;
        if (k0 == Cin) { k0 = 0; ++tap; }
    };
    auto store_tile = [&](int pp) {
        for (int s = 0; s < 2; ++s) *(v8bf*)&AsB[pp][ar[s] * LDSW + ac[s]] = ra[s];
        for (int s = 0; s < 2; ++s)
            for (int i = 0; i < 8; ++i)
                BsB[pp][(bn[s] + i) * LDSW + bk[s]] = rbv[s * 8 + i];
    };

    load_tile();
    store_tile(0);
    int p = 0;
    for (int it = 0; it < steps; ++it) {
        const bool hasNext = (it + 1 < steps);
        if (hasNext) load_tile();     // overlaps with this tile's WMMA block
        __syncthreads();
        const bf16* As = AsB[p];
        const bf16* Bs = BsB[p];
        BFrag a0, a1, b0, b1;
        a0.h[0] = *(const v8bf*)&As[(mh + fr) * LDSW + kb];
        a0.h[1] = *(const v8bf*)&As[(mh + fr) * LDSW + kb + 16];
        a1.h[0] = *(const v8bf*)&As[(mh + 16 + fr) * LDSW + kb];
        a1.h[1] = *(const v8bf*)&As[(mh + 16 + fr) * LDSW + kb + 16];
        b0.h[0] = *(const v8bf*)&Bs[(nh + fr) * LDSW + kb];
        b0.h[1] = *(const v8bf*)&Bs[(nh + fr) * LDSW + kb + 16];
        b1.h[0] = *(const v8bf*)&Bs[(nh + 16 + fr) * LDSW + kb];
        b1.h[1] = *(const v8bf*)&Bs[(nh + 16 + fr) * LDSW + kb + 16];
        acc00 = __builtin_amdgcn_wmma_f32_16x16x32_bf16(false, a0.v, false, b0.v,
                                                        (short)0, acc00, false, false);
        acc01 = __builtin_amdgcn_wmma_f32_16x16x32_bf16(false, a0.v, false, b1.v,
                                                        (short)0, acc01, false, false);
        acc10 = __builtin_amdgcn_wmma_f32_16x16x32_bf16(false, a1.v, false, b0.v,
                                                        (short)0, acc10, false, false);
        acc11 = __builtin_amdgcn_wmma_f32_16x16x32_bf16(false, a1.v, false, b1.v,
                                                        (short)0, acc11, false, false);
        if (hasNext) store_tile(p ^ 1);
        p ^= 1;
    }
    for (int q = 0; q < 4; ++q) {
        v8f av = (q == 0) ? acc00 : (q == 1) ? acc01 : (q == 2) ? acc10 : acc11;
        int mb = mh + ((q >> 1) & 1) * 16;
        int n = n0 + nh + (q & 1) * 16 + fr;
        for (int r = 0; r < 8; ++r) {
            int m = m0 + mb + ((lane < 16) ? r : r + 8);
            epi_store(e, bz, m, n, av[r]);
        }
    }
}

// ------------------------------ helper kernels -----------------------------
__global__ void cvt_f32_bf16(const float* __restrict__ s, bf16* __restrict__ d,
                             long long n) {
    long long i = (long long)blockIdx.x * blockDim.x + threadIdx.x;
    long long st = (long long)gridDim.x * blockDim.x;
    for (; i < n; i += st) d[i] = (bf16)s[i];
}

// OIHW (3x3) -> [tap][Cout][Cin] bf16
__global__ void repack3x3(const float* __restrict__ w, bf16* __restrict__ wp,
                          int Cout, int Cin) {
    long long total = (long long)Cout * Cin * 9;
    for (long long i = (long long)blockIdx.x * blockDim.x + threadIdx.x; i < total;
         i += (long long)gridDim.x * blockDim.x) {
        int ci = (int)(i % Cin);
        long long t = i / Cin;
        int co = (int)(t % Cout), tap = (int)(t / Cout);
        wp[i] = (bf16)w[((long long)co * Cin + ci) * 9 + tap];
    }
}

// row softmax; flip=1 -> softmax(rowmax - e) == exp(rowmin - e)/sum
__global__ void softmax_rows(const float* __restrict__ E, float* outF, bf16* outB,
                             int cols, int flip) {
    const long long base = (long long)blockIdx.x * cols;
    __shared__ float red[256];
    float lmax = -3.4e38f, lmin = 3.4e38f;
    for (int i = threadIdx.x; i < cols; i += 256) {
        float v = E[base + i];
        lmax = fmaxf(lmax, v); lmin = fminf(lmin, v);
    }
    red[threadIdx.x] = lmax; __syncthreads();
    for (int s = 128; s > 0; s >>= 1) {
        if (threadIdx.x < s) red[threadIdx.x] = fmaxf(red[threadIdx.x], red[threadIdx.x + s]);
        __syncthreads();
    }
    float rmax = red[0]; __syncthreads();
    red[threadIdx.x] = lmin; __syncthreads();
    for (int s = 128; s > 0; s >>= 1) {
        if (threadIdx.x < s) red[threadIdx.x] = fminf(red[threadIdx.x], red[threadIdx.x + s]);
        __syncthreads();
    }
    float rmin = red[0]; __syncthreads();
    const float ref = flip ? rmin : rmax;
    const float sgn = flip ? -1.f : 1.f;
    float ls = 0.f;
    for (int i = threadIdx.x; i < cols; i += 256) ls += __expf(sgn * (E[base + i] - ref));
    red[threadIdx.x] = ls; __syncthreads();
    for (int s = 128; s > 0; s >>= 1) {
        if (threadIdx.x < s) red[threadIdx.x] += red[threadIdx.x + s];
        __syncthreads();
    }
    const float inv = 1.f / red[0];
    for (int i = threadIdx.x; i < cols; i += 256) {
        float z = __expf(sgn * (E[base + i] - ref)) * inv;
        if (outF) outF[base + i] = z;
        if (outB) outB[base + i] = (bf16)z;
    }
}

// mean over last dim (bf16 input), one block per row
__global__ void gap_rows(const bf16* __restrict__ X, float* __restrict__ out, int n) {
    const long long base = (long long)blockIdx.x * n;
    __shared__ float red[256];
    float s = 0.f;
    for (int i = threadIdx.x; i < n; i += 256) s += (float)X[base + i];
    red[threadIdx.x] = s; __syncthreads();
    for (int k = 128; k > 0; k >>= 1) {
        if (threadIdx.x < k) red[threadIdx.x] += red[threadIdx.x + k];
        __syncthreads();
    }
    if (threadIdx.x == 0) out[blockIdx.x] = red[0] / (float)n;
}

// small dense: out[b,m] = act(scale[m]*dot(W[m,:],in[b,:]) + bias[m])
__global__ void gemv_small(const float* __restrict__ W, const float* __restrict__ in,
                           const float* scale, const float* bias,
                           float* __restrict__ out, int M, int K, int Btot, int act) {
    int idx = blockIdx.x * blockDim.x + threadIdx.x;
    if (idx >= Btot * M) return;
    int bb = idx / M, m = idx % M;
    const float* w = W + (long long)m * K;
    const float* x = in + (long long)bb * K;
    float s = 0.f;
    for (int k = 0; k < K; ++k) s += w[k] * x[k];
    if (scale) s *= scale[m];
    if (bias)  s += bias[m];
    if (act == 1) s = s > 0.f ? s : 0.f;
    else if (act == 2) s = 1.f / (1.f + __expf(-s));
    out[idx] = s;
}

// concat[b, C+c, n] = bf16((1 + s2[b,c]) * bottle[b,c,n])
__global__ void se_combine(const float* __restrict__ bottle, const float* __restrict__ s2,
                           bf16* __restrict__ cat, int C, int Npix, long long cBatch,
                           long long total) {
    for (long long i = (long long)blockIdx.x * blockDim.x + threadIdx.x; i < total;
         i += (long long)gridDim.x * blockDim.x) {
        int n = (int)(i % Npix);
        long long t = i / Npix;
        int c = (int)(t % C);
        int b = (int)(t / C);
        float f = (1.f + s2[b * C + c]) * bottle[i];
        cat[(long long)b * cBatch + (long long)(C + c) * Npix + n] = (bf16)f;
    }
}

// dst[b, rowOff+c, n] = bf16(pool[b,c])   (broadcast of 1x1 pooled feature)
__global__ void bcast_pool(const float* __restrict__ pool, bf16* __restrict__ dst,
                           int C, int Npix, long long dBatch, int rowOff,
                           long long total) {
    for (long long i = (long long)blockIdx.x * blockDim.x + threadIdx.x; i < total;
         i += (long long)gridDim.x * blockDim.x) {
        int n = (int)(i % Npix);
        long long t = i / Npix;
        int c = (int)(t % C);
        int b = (int)(t / C);
        dst[(long long)b * dBatch + (long long)(rowOff + c) * Npix + n] = (bf16)pool[b * C + c];
    }
}

// bilinear align_corners upsample; dst = add + gamma*bilin(src) (add/gamma optional)
__global__ void upsample_bilin(const float* __restrict__ src, float* __restrict__ dst,
                               const float* addp, const float* gammap,
                               int C, int Hi, int Wi, int Ho, int Wo, long long total) {
    for (long long i = (long long)blockIdx.x * blockDim.x + threadIdx.x; i < total;
         i += (long long)gridDim.x * blockDim.x) {
        int wo = (int)(i % Wo);
        long long t = i / Wo;
        int ho = (int)(t % Ho); t /= Ho;
        int c  = (int)(t % C);
        int b  = (int)(t / C);
        float sy = (Ho > 1) ? (float)ho * (float)(Hi - 1) / (float)(Ho - 1) : 0.f;
        float sx = (Wo > 1) ? (float)wo * (float)(Wi - 1) / (float)(Wo - 1) : 0.f;
        int y0 = (int)sy; if (y0 > Hi - 2) y0 = Hi - 2; if (y0 < 0) y0 = 0;
        int x0 = (int)sx; if (x0 > Wi - 2) x0 = Wi - 2; if (x0 < 0) x0 = 0;
        float fy = sy - (float)y0, fx = sx - (float)x0;
        const float* s = src + ((long long)(b * C + c) * Hi) * Wi;
        float v00 = s[y0 * Wi + x0],       v01 = s[y0 * Wi + x0 + 1];
        float v10 = s[(y0 + 1) * Wi + x0], v11 = s[(y0 + 1) * Wi + x0 + 1];
        float v = (1.f - fy) * ((1.f - fx) * v00 + fx * v01)
                + fy * ((1.f - fx) * v10 + fx * v11);
        float g = gammap ? gammap[0] : 1.f;
        dst[i] = (addp ? addp[i] : 0.f) + g * v;
    }
}

// final classifier conv (M=19 too small for WMMA tiles)
__global__ void conv1x1_naive(const float* __restrict__ W, const float* __restrict__ bias,
                              const float* __restrict__ X, float* __restrict__ out,
                              int M, int Cin, int Npix, long long total) {
    for (long long i = (long long)blockIdx.x * blockDim.x + threadIdx.x; i < total;
         i += (long long)gridDim.x * blockDim.x) {
        int n = (int)(i % Npix);
        long long t = i / Npix;
        int m = (int)(t % M);
        int b = (int)(t / M);
        const float* x = X + (long long)b * Cin * Npix + n;
        float s = bias[m];
        const float* w = W + (long long)m * Cin;
        for (int c = 0; c < Cin; ++c) s += w[c] * x[(long long)c * Npix];
        out[i] = s;
    }
}

// ------------------------------- host side ---------------------------------
static inline Epi epi0() {
    Epi e; e.scale = nullptr; e.bias = nullptr; e.gamma = nullptr;
    e.res = nullptr; e.resBatch = 0; e.ldRes = 0; e.relu = 0;
    e.outF = nullptr; e.outFBatch = 0; e.ldF = 0;
    e.outB = nullptr; e.outBBatch = 0; e.ldBq = 0;
    return e;
}

static inline void* take(char*& p, size_t bytes) {
    void* r = p; p += (bytes + 255) & ~(size_t)255; return r;
}

static inline void cvtK(hipStream_t st, const float* s, bf16* d, long long n) {
    long long blocks = (n + 2047) / 2048;
    if (blocks > 16384) blocks = 16384; if (blocks < 1) blocks = 1;
    cvt_f32_bf16<<<(int)blocks, 256, 0, st>>>(s, d, n);
}

static inline void gemm(hipStream_t st, const bf16* A, long long aB, int ldA, int tA,
                        const bf16* Bm, long long bB, int ldB, int tB,
                        int M, int N, int K, int batch, Epi e) {
    dim3 g(N / 64, M / 64, batch);
    gemm_wmma<<<g, 128, 0, st>>>(A, aB, ldA, tA, Bm, bB, ldB, tB, M, N, K, e);
}

static inline int ilog2(int v) { int s = 0; while ((1 << s) < v) ++s; return s; }

static inline void conv(hipStream_t st, const bf16* X, const bf16* Wp,
                        int Cin, int Cout, int Hin, int Win, int Hout, int Wout,
                        int stride, int pad, int dil, int batch, Epi e) {
    dim3 g((Hout * Wout) / 64, Cout / 64, batch);
    conv3x3_wmma<<<g, 128, 0, st>>>(X, Wp, Cin, Cout, Hin, Win, Wout, ilog2(Wout),
                                    stride, pad, dil, e);
}

// param-index layout (insertion order of the reference param dict)
enum {
    IN_X = 0,
    IN_B0 = 1, IN_B1 = 4, IN_B2 = 7, IN_B3 = 10, IN_B4 = 13,
    IN_POOL2 = 16, IN_POOL3 = 19, IN_POOL4 = 22,
    IN_PAM2 = 25, IN_PAM3 = 32, IN_PAM4 = 39,
    IN_PAP_GAMMA = 46,
    IN_GSC0 = 47, IN_GSC1 = 61, IN_GSC2 = 75, IN_GSC3 = 89,
    IN_GSC_TOP = 103, IN_SEC = 117,
    IN_CONV8_W = 131, IN_CONV8_B = 132
};
// cbr: +0 w, +1 s, +2 b
// pam: +0 qw +1 qb +2 kw +3 kb +4 vw +5 vb +6 gamma
// gsc: +0 qcw +1 qcs +2 qcb +3 gamma +4 projw +5 projs +6 projb
//      +7 se1w +8 se1b +9 se2w +10 se2b +11 fusew +12 fuses +13 fuseb

struct PamS {
    bf16 *qw, *kw, *vw;
    bf16 *q, *k;
    float* vf; bf16* vbf;
    float* energy; bf16* attnb;
};

static void run_pam(hipStream_t st, void* const* d_in, int base, const bf16* dbf,
                    int Ns, float* pout, const PamS& S) {
    auto P = [&](int i) { return (const float*)d_in[i]; };
    const int Bn = 4, KD = 64, OC = 256;
    cvtK(st, P(base + 0), S.qw, (long long)KD * OC);
    cvtK(st, P(base + 2), S.kw, (long long)KD * OC);
    cvtK(st, P(base + 4), S.vw, (long long)OC * OC);
    { Epi e = epi0(); e.bias = P(base + 1);
      e.outB = S.q; e.outBBatch = (long long)KD * Ns; e.ldBq = Ns;
      gemm(st, S.qw, 0, OC, 0, dbf, (long long)OC * Ns, Ns, 0, KD, Ns, OC, Bn, e); }
    { Epi e = epi0(); e.bias = P(base + 3);
      e.outB = S.k; e.outBBatch = (long long)KD * Ns; e.ldBq = Ns;
      gemm(st, S.kw, 0, OC, 0, dbf, (long long)OC * Ns, Ns, 0, KD, Ns, OC, Bn, e); }
    { Epi e = epi0(); e.bias = P(base + 5);
      e.outF = S.vf; e.outFBatch = (long long)OC * Ns; e.ldF = Ns;
      e.outB = S.vbf; e.outBBatch = (long long)OC * Ns; e.ldBq = Ns;
      gemm(st, S.vw, 0, OC, 0, dbf, (long long)OC * Ns, Ns, 0, OC, Ns, OC, Bn, e); }
    // energy[n,m] = sum_k q[k,n] * k[k,m]
    { Epi e = epi0(); e.outF = S.energy; e.outFBatch = (long long)Ns * Ns; e.ldF = Ns;
      gemm(st, S.q, (long long)KD * Ns, Ns, 1, S.k, (long long)KD * Ns, Ns, 0,
           Ns, Ns, KD, Bn, e); }
    softmax_rows<<<Bn * Ns, 256, 0, st>>>(S.energy, nullptr, S.attnb, Ns, 0);
    // out[c,n] = gamma * sum_m v[c,m]*attn[n,m] + v[c,n]
    { Epi e = epi0(); e.gamma = P(base + 6);
      e.res = S.vf; e.resBatch = (long long)OC * Ns; e.ldRes = Ns;
      e.outF = pout; e.outFBatch = (long long)OC * Ns; e.ldF = Ns;
      gemm(st, S.vbf, (long long)OC * Ns, Ns, 0, S.attnb, (long long)Ns * Ns, Ns, 1,
           OC, Ns, Ns, Bn, e); }
}

struct GscS {
    bf16 *qcwb, *projwb, *fusewb;
    float* qf; bf16* qbf;
    float* energy; bf16* attnb;
    float* bottle; bf16* cat;
    float* mean; float* s1; float* s2;
};

static void run_gsc(hipStream_t st, void* const* d_in, int base, const bf16* Fbf,
                    int Cin, float* outF, const float* resF, bf16* outB,
                    long long outBBatch, const GscS& S) {
    auto P = [&](int i) { return (const float*)d_in[i]; };
    const int Bn = 4, HW = 4096, OC = 256;
    const int M1 = Cin / 16;
    cvtK(st, P(base + 0), S.qcwb, (long long)OC * Cin);
    cvtK(st, P(base + 4), S.projwb, (long long)OC * Cin);
    cvtK(st, P(base + 11), S.fusewb, (long long)OC * 512);
    // q = relu(bn(conv1x1))
    { Epi e = epi0(); e.scale = P(base + 1); e.bias = P(base + 2); e.relu = 1;
      e.outF = S.qf; e.outFBatch = (long long)OC * HW; e.ldF = HW;
      e.outB = S.qbf; e.outBBatch = (long long)OC * HW; e.ldBq = HW;
      gemm(st, S.qcwb, 0, Cin, 0, Fbf, (long long)Cin * HW, HW, 0, OC, HW, Cin, Bn, e); }
    // energy[q,c] = sum_n qf[q,n]*F[c,n]
    { Epi e = epi0(); e.outF = S.energy; e.outFBatch = (long long)OC * Cin; e.ldF = Cin;
      gemm(st, S.qbf, (long long)OC * HW, HW, 0, Fbf, (long long)Cin * HW, HW, 1,
           OC, Cin, HW, Bn, e); }
    softmax_rows<<<Bn * OC, 256, 0, st>>>(S.energy, nullptr, S.attnb, Cin, 1);
    // g = gamma * attn x F + q   -> concat rows [0,256)
    { Epi e = epi0(); e.gamma = P(base + 3);
      e.res = S.qf; e.resBatch = (long long)OC * HW; e.ldRes = HW;
      e.outB = S.cat; e.outBBatch = 512LL * HW; e.ldBq = HW;
      gemm(st, S.attnb, (long long)OC * Cin, Cin, 0, Fbf, (long long)Cin * HW, HW, 0,
           OC, HW, Cin, Bn, e); }
    // bottle = relu(bn(proj))
    { Epi e = epi0(); e.scale = P(base + 5); e.bias = P(base + 6); e.relu = 1;
      e.outF = S.bottle; e.outFBatch = (long long)OC * HW; e.ldF = HW;
      gemm(st, S.projwb, 0, Cin, 0, Fbf, (long long)Cin * HW, HW, 0, OC, HW, Cin, Bn, e); }
    // SE branch
    gap_rows<<<Bn * Cin, 256, 0, st>>>(Fbf, S.mean, HW);
    gemv_small<<<(Bn * M1 + 255) / 256, 256, 0, st>>>(P(base + 7), S.mean, nullptr,
                                                      P(base + 8), S.s1, M1, Cin, Bn, 1);
    gemv_small<<<(Bn * OC + 255) / 256, 256, 0, st>>>(P(base + 9), S.s1, nullptr,
                                                      P(base + 10), S.s2, OC, M1, Bn, 2);
    se_combine<<<8192, 256, 0, st>>>(S.bottle, S.s2, S.cat, OC, HW, 512LL * HW,
                                     (long long)Bn * OC * HW);
    // fuse = relu(bn(conv1x1(concat)))  [+ resF for the final 'sec' branch]
    { Epi e = epi0(); e.scale = P(base + 12); e.bias = P(base + 13); e.relu = 1;
      if (resF) { e.res = resF; e.resBatch = (long long)OC * HW; e.ldRes = HW; }
      if (outF) { e.outF = outF; e.outFBatch = (long long)OC * HW; e.ldF = HW; }
      if (outB) { e.outB = outB; e.outBBatch = outBBatch; e.ldBq = HW; }
      gemm(st, S.fusewb, 0, 512, 0, S.cat, 512LL * HW, HW, 0, OC, HW, 512, Bn, e); }
}

extern "C" void kernel_launch(void* const* d_in, const int* in_sizes, int n_in,
                              void* d_out, int out_size, void* d_ws, size_t ws_size,
                              hipStream_t stream) {
    (void)in_sizes; (void)n_in; (void)out_size; (void)ws_size;
    auto P = [&](int i) { return (const float*)d_in[i]; };
    const int Bn = 4, IC = 2048, OC = 256, HW = 4096;
    char* wp = (char*)d_ws;

    // ---- workspace carve (~390 MB) ----
    bf16*  xb     = (bf16*)take(wp, (size_t)Bn * IC * HW * 2);
    bf16*  w1p    = (bf16*)take(wp, (size_t)9 * OC * IC * 2);
    bf16*  w2p    = (bf16*)take(wp, (size_t)9 * OC * IC * 2);
    bf16*  w3p    = (bf16*)take(wp, (size_t)9 * OC * IC * 2);
    bf16*  wp2p   = (bf16*)take(wp, (size_t)9 * OC * OC * 2);
    bf16*  wp3p   = (bf16*)take(wp, (size_t)9 * OC * OC * 2);
    bf16*  wp4p   = (bf16*)take(wp, (size_t)9 * OC * OC * 2);
    bf16*  wb0    = (bf16*)take(wp, (size_t)OC * IC * 2);
    float* f0in   = (float*)take(wp, (size_t)Bn * OC * HW * 4);
    bf16*  f0inb  = (bf16*)take(wp, (size_t)Bn * OC * HW * 2);
    bf16*  d2b    = (bf16*)take(wp, (size_t)Bn * OC * 1024 * 2);
    bf16*  d3b    = (bf16*)take(wp, (size_t)Bn * OC * 256 * 2);
    bf16*  d4b    = (bf16*)take(wp, (size_t)Bn * OC * 64 * 2);
    PamS pamS;
    pamS.qw     = (bf16*)take(wp, (size_t)64 * OC * 2);
    pamS.kw     = (bf16*)take(wp, (size_t)64 * OC * 2);
    pamS.vw     = (bf16*)take(wp, (size_t)OC * OC * 2);
    pamS.q      = (bf16*)take(wp, (size_t)Bn * 64 * 1024 * 2);
    pamS.k      = (bf16*)take(wp, (size_t)Bn * 64 * 1024 * 2);
    pamS.vf     = (float*)take(wp, (size_t)Bn * OC * 1024 * 4);
    pamS.vbf    = (bf16*)take(wp, (size_t)Bn * OC * 1024 * 2);
    pamS.energy = (float*)take(wp, (size_t)Bn * 1024 * 1024 * 4);
    pamS.attnb  = (bf16*)take(wp, (size_t)Bn * 1024 * 1024 * 2);
    float* pout2 = (float*)take(wp, (size_t)Bn * OC * 1024 * 4);
    float* pout3 = (float*)take(wp, (size_t)Bn * OC * 256 * 4);
    float* pout4 = (float*)take(wp, (size_t)Bn * OC * 64 * 4);
    float* u4buf = (float*)take(wp, (size_t)Bn * OC * 256 * 4);
    float* u3buf = (float*)take(wp, (size_t)Bn * OC * 1024 * 4);
    float* papout  = (float*)take(wp, (size_t)Bn * OC * HW * 4);
    bf16*  papoutb = (bf16*)take(wp, (size_t)Bn * OC * HW * 2);
    bf16*  fb1   = (bf16*)take(wp, (size_t)Bn * OC * HW * 2);
    bf16*  fb2   = (bf16*)take(wp, (size_t)Bn * OC * HW * 2);
    bf16*  fb3   = (bf16*)take(wp, (size_t)Bn * OC * HW * 2);
    bf16*  catTop = (bf16*)take(wp, (size_t)Bn * 1280 * HW * 2);
    float* ytop  = (float*)take(wp, (size_t)Bn * OC * HW * 4);
    float* feat  = (float*)take(wp, (size_t)Bn * OC * HW * 4);
    float* xmean = (float*)take(wp, (size_t)Bn * IC * 4);
    float* poolv = (float*)take(wp, (size_t)Bn * OC * 4);
    GscS S;
    S.qcwb   = (bf16*)take(wp, (size_t)OC * IC * 2);
    S.projwb = (bf16*)take(wp, (size_t)OC * IC * 2);
    S.fusewb = (bf16*)take(wp, (size_t)OC * 512 * 2);
    S.qf     = (float*)take(wp, (size_t)Bn * OC * HW * 4);
    S.qbf    = (bf16*)take(wp, (size_t)Bn * OC * HW * 2);
    S.energy = (float*)take(wp, (size_t)Bn * OC * IC * 4);
    S.attnb  = (bf16*)take(wp, (size_t)Bn * OC * IC * 2);
    S.bottle = (float*)take(wp, (size_t)Bn * OC * HW * 4);
    S.cat    = (bf16*)take(wp, (size_t)Bn * 512 * HW * 2);
    S.mean   = (float*)take(wp, (size_t)Bn * IC * 4);
    S.s1     = (float*)take(wp, (size_t)Bn * 128 * 4);
    S.s2     = (float*)take(wp, (size_t)Bn * OC * 4);

    // ---- stage 0: precision conversion / weight repack ----
    cvtK(stream, P(IN_X), xb, (long long)Bn * IC * HW);
    repack3x3<<<4096, 256, 0, stream>>>(P(IN_B1), w1p, OC, IC);
    repack3x3<<<4096, 256, 0, stream>>>(P(IN_B2), w2p, OC, IC);
    repack3x3<<<4096, 256, 0, stream>>>(P(IN_B3), w3p, OC, IC);
    repack3x3<<<2048, 256, 0, stream>>>(P(IN_POOL2), wp2p, OC, OC);
    repack3x3<<<2048, 256, 0, stream>>>(P(IN_POOL3), wp3p, OC, OC);
    repack3x3<<<2048, 256, 0, stream>>>(P(IN_POOL4), wp4p, OC, OC);
    cvtK(stream, P(IN_B0), wb0, (long long)OC * IC);

    // ---- b0: f0in = relu(bn(conv1x1(x))) ----
    { Epi e = epi0(); e.scale = P(IN_B0 + 1); e.bias = P(IN_B0 + 2); e.relu = 1;
      e.outF = f0in; e.outFBatch = (long long)OC * HW; e.ldF = HW;
      e.outB = f0inb; e.outBBatch = (long long)OC * HW; e.ldBq = HW;
      gemm(stream, wb0, 0, IC, 0, xb, (long long)IC * HW, HW, 0, OC, HW, IC, Bn, e); }

    // ---- PAP: strided pool chain + PAMs + upsample pyramid ----
    { Epi e = epi0(); e.scale = P(IN_POOL2 + 1); e.bias = P(IN_POOL2 + 2); e.relu = 1;
      e.outB = d2b; e.outBBatch = (long long)OC * 1024; e.ldBq = 1024;
      conv(stream, f0inb, wp2p, OC, OC, 64, 64, 32, 32, 2, 1, 1, Bn, e); }
    { Epi e = epi0(); e.scale = P(IN_POOL3 + 1); e.bias = P(IN_POOL3 + 2); e.relu = 1;
      e.outB = d3b; e.outBBatch = (long long)OC * 256; e.ldBq = 256;
      conv(stream, d2b, wp3p, OC, OC, 32, 32, 16, 16, 2, 1, 1, Bn, e); }
    { Epi e = epi0(); e.scale = P(IN_POOL4 + 1); e.bias = P(IN_POOL4 + 2); e.relu = 1;
      e.outB = d4b; e.outBBatch = (long long)OC * 64; e.ldBq = 64;
      conv(stream, d3b, wp4p, OC, OC, 16, 16, 8, 8, 2, 1, 1, Bn, e); }
    run_pam(stream, d_in, IN_PAM2, d2b, 1024, pout2, pamS);
    run_pam(stream, d_in, IN_PAM3, d3b, 256, pout3, pamS);
    run_pam(stream, d_in, IN_PAM4, d4b, 64, pout4, pamS);
    { long long t = (long long)Bn * OC * 256;
      upsample_bilin<<<(int)((t + 255) / 256), 256, 0, stream>>>(
          pout4, u4buf, pout3, nullptr, OC, 8, 8, 16, 16, t); }
    { long long t = (long long)Bn * OC * 1024;
      upsample_bilin<<<(int)((t + 255) / 256), 256, 0, stream>>>(
          u4buf, u3buf, pout2, nullptr, OC, 16, 16, 32, 32, t); }
    { long long t = (long long)Bn * OC * HW;
      upsample_bilin<<<8192, 256, 0, stream>>>(
          u3buf, papout, f0in, P(IN_PAP_GAMMA), OC, 32, 32, 64, 64, t); }
    cvtK(stream, papout, papoutb, (long long)Bn * OC * HW);

    // ---- dilated ASPP branches b1..b3 ----
    { Epi e = epi0(); e.scale = P(IN_B1 + 1); e.bias = P(IN_B1 + 2); e.relu = 1;
      e.outB = fb1; e.outBBatch = (long long)OC * HW; e.ldBq = HW;
      conv(stream, xb, w1p, IC, OC, 64, 64, 64, 64, 1, 12, 12, Bn, e); }
    { Epi e = epi0(); e.scale = P(IN_B2 + 1); e.bias = P(IN_B2 + 2); e.relu = 1;
      e.outB = fb2; e.outBBatch = (long long)OC * HW; e.ldBq = HW;
      conv(stream, xb, w2p, IC, OC, 64, 64, 64, 64, 1, 24, 24, Bn, e); }
    { Epi e = epi0(); e.scale = P(IN_B3 + 1); e.bias = P(IN_B3 + 2); e.relu = 1;
      e.outB = fb3; e.outBBatch = (long long)OC * HW; e.ldBq = HW;
      conv(stream, xb, w3p, IC, OC, 64, 64, 64, 64, 1, 36, 36, Bn, e); }

    // ---- per-branch GSC, results written straight into the 1280-ch concat ----
    const long long catB = 1280LL * HW;
    run_gsc(stream, d_in, IN_GSC0, papoutb, OC, nullptr, nullptr, catTop, catB, S);
    run_gsc(stream, d_in, IN_GSC1, fb1, OC, nullptr, nullptr, catTop + 256LL * HW, catB, S);
    run_gsc(stream, d_in, IN_GSC2, fb2, OC, nullptr, nullptr, catTop + 512LL * HW, catB, S);
    run_gsc(stream, d_in, IN_GSC3, fb3, OC, nullptr, nullptr, catTop + 768LL * HW, catB, S);

    // ---- global-pool branch (b4) broadcast into concat rows [1024,1280) ----
    gap_rows<<<Bn * IC, 256, 0, stream>>>(xb, xmean, HW);
    gemv_small<<<(Bn * OC + 255) / 256, 256, 0, stream>>>(
        P(IN_B4), xmean, P(IN_B4 + 1), P(IN_B4 + 2), poolv, OC, IC, Bn, 1);
    bcast_pool<<<8192, 256, 0, stream>>>(poolv, catTop, OC, HW, catB, 1024,
                                         (long long)Bn * OC * HW);

    // ---- top GSC over the 1280-channel concat ----
    run_gsc(stream, d_in, IN_GSC_TOP, catTop, 1280, ytop, nullptr, nullptr, 0, S);

    // ---- secondary GSC on raw x; fuse epilogue adds ytop -> feat ----
    run_gsc(stream, d_in, IN_SEC, xb, IC, feat, ytop, nullptr, 0, S);

    // ---- classifier conv8 (19 classes) ----
    { long long t = (long long)Bn * 19 * HW;
      conv1x1_naive<<<(int)((t + 255) / 256), 256, 0, stream>>>(
          P(IN_CONV8_W), P(IN_CONV8_B), feat, (float*)d_out, 19, OC, HW, t); }
}
